// TransformerGroupQueryAttention_54726473286127
// MI455X (gfx1250) — matrix operand order repaired and checked by reading, not compiled, not run on hardware
//
#include <hip/hip_runtime.h>
#include <hip/hip_bf16.h>

// ---- problem constants (from reference) ----
#define BB   2
#define SS   1024
#define DD   4096
#define HH   32
#define KVHH 8
#define HDD  128
#define GROUPP 4
#define SCALE_QK 0.08838834764831845f  // 1/sqrt(128)

typedef __attribute__((ext_vector_type(16))) __bf16 v16bf;
typedef __attribute__((ext_vector_type(8)))  __bf16 bf16x8;
typedef __attribute__((ext_vector_type(8)))  float  v8f;
typedef __attribute__((ext_vector_type(4)))  float  f32x4;

#define WMMA_BF16(Af, Bf, Cf) \
  __builtin_amdgcn_wmma_f32_16x16x32_bf16(false, (Af), false, (Bf), (short)0, (Cf), false, false)

// B-operand fragment (K=32): lanes 0-15 hold K=0..15, lanes 16-31 hold K=16..31.
// Caller passes pointer already offset by (lo?0:16); 16 consecutive bf16 = 32 B.
__device__ inline v16bf load_b_frag_bf(const __bf16* __restrict__ p) {
  return *(const v16bf*)p;
}

// A-operand fragment (16-bit A, 16x32): lane<16 -> row M=lane, elems 0..15 = K 0..15 ;
// lane>=16 -> row M=lane-16, elems 0..15 = K 16..31.
__device__ inline v16bf load_a_frag_bf(const __bf16* __restrict__ rowp, bool lo) {
  bf16x8 u0 = *(const bf16x8*)(rowp + (lo ? 0 : 16));
  bf16x8 u1 = *(const bf16x8*)(rowp + 8 + (lo ? 0 : 16));
  v16bf r;
#pragma unroll
  for (int i = 0; i < 8; ++i) { r[i] = u0[i]; r[8 + i] = u1[i]; }
  return r;
}

// Same A layout sourced from an LDS row of bf16 bits.
__device__ inline v16bf load_a_frag_lds(const unsigned short* __restrict__ rowp, bool lo) {
  bf16x8 u0 = *(const bf16x8*)(rowp + (lo ? 0 : 16));
  bf16x8 u1 = *(const bf16x8*)(rowp + 8 + (lo ? 0 : 16));
  v16bf r;
#pragma unroll
  for (int i = 0; i < 8; ++i) { r[i] = u0[i]; r[8 + i] = u1[i]; }
  return r;
}

// ---------------------------------------------------------------------------
// One-time fp32 -> bf16 conversion (8 elements/thread, packed 16-B stores).
// ---------------------------------------------------------------------------
__global__ __launch_bounds__(256) void gqa_cvt_bf16(
    const float* __restrict__ src, __bf16* __restrict__ dst, int n)
{
  int t = blockIdx.x * blockDim.x + threadIdx.x;
  int i = t * 8;
  if (i >= n) return;
  f32x4 a = *(const f32x4*)(src + i);
  f32x4 b = *(const f32x4*)(src + i + 4);
  bf16x8 r;
#pragma unroll
  for (int j = 0; j < 4; ++j) { r[j] = (__bf16)a[j]; r[4 + j] = (__bf16)b[j]; }
  *(bf16x8*)(dst + i) = r;
}

// ---------------------------------------------------------------------------
// GEMM: C[M,N] = A[M,K] @ W[N,K]^T   (bf16 in, OutT out, bf16 WMMA math)
// block = 256 threads = 8 waves; block tile 128(M) x 64(N);
// wave (wm=w&3, wn=w>>2) owns a 32x32 sub-tile -> 4 accumulators,
// 4 WMMAs per 8 x b128 loads per 32-K chunk.
// ---------------------------------------------------------------------------
template <typename OutT>
__global__ __launch_bounds__(256) void gqa_gemm_wt(
    const __bf16* __restrict__ A, const __bf16* __restrict__ W,
    OutT* __restrict__ C, int M, int N, int K)
{
  const int lane = threadIdx.x & 31;
  const int wave = threadIdx.x >> 5;
  const bool lo  = lane < 16;
  const int  lq  = lane & 15;
  const int m_base = blockIdx.y * 128 + (wave & 3) * 32;
  const int n_base = blockIdx.x * 64  + (wave >> 2) * 32;

  const __bf16* arow0 = A + (size_t)(m_base + lq) * K;
  const __bf16* arow1 = A + (size_t)(m_base + 16 + lq) * K;
  const __bf16* wrow0 = W + (size_t)(n_base + lq) * K;
  const __bf16* wrow1 = W + (size_t)(n_base + 16 + lq) * K;

  v8f acc00 = {}, acc01 = {}, acc10 = {}, acc11 = {};
  for (int kc = 0; kc < K; kc += 32) {
    __builtin_prefetch(arow0 + kc + 256, 0, 0);   // global_prefetch_b8
    __builtin_prefetch(wrow0 + kc + 256, 0, 0);
    v16bf a0 = load_a_frag_bf(arow0 + kc, lo);
    v16bf a1 = load_a_frag_bf(arow1 + kc, lo);
    v16bf b0 = load_b_frag_bf(wrow0 + kc + (lo ? 0 : 16));
    v16bf b1 = load_b_frag_bf(wrow1 + kc + (lo ? 0 : 16));
    acc00 = WMMA_BF16(a0, b0, acc00);
    acc01 = WMMA_BF16(a0, b1, acc01);
    acc10 = WMMA_BF16(a1, b0, acc10);
    acc11 = WMMA_BF16(a1, b1, acc11);
  }
  // C/D layout: VGPR j -> row +(lo? j : 8+j), lane -> column.
  const int mrow0 = m_base + (lo ? 0 : 8);
  const int mrow1 = mrow0 + 16;
#pragma unroll
  for (int j = 0; j < 8; ++j) {
    C[(size_t)(mrow0 + j) * N + n_base + lq]      = (OutT)acc00[j];
    C[(size_t)(mrow0 + j) * N + n_base + 16 + lq] = (OutT)acc01[j];
    C[(size_t)(mrow1 + j) * N + n_base + lq]      = (OutT)acc10[j];
    C[(size_t)(mrow1 + j) * N + n_base + 16 + lq] = (OutT)acc11[j];
  }
}

// ---------------------------------------------------------------------------
// RoPE in place on bf16 [B,S,NH,HD]; one thread per (even,odd) pair.
// ---------------------------------------------------------------------------
__global__ __launch_bounds__(256) void gqa_rope(
    __bf16* __restrict__ x, const float* __restrict__ fcos,
    const float* __restrict__ fsin, int NH)
{
  size_t t = (size_t)blockIdx.x * blockDim.x + threadIdx.x;
  size_t total = (size_t)BB * SS * NH * (HDD / 2);
  if (t >= total) return;
  int p  = (int)(t & 63); size_t r = t >> 6;
  int hh = (int)(r % NH); r /= NH;
  int s  = (int)(r % SS);
  int b  = (int)(r / SS);
  size_t base = (((size_t)(b * SS + s)) * NH + hh) * HDD + 2 * p;
  float c  = fcos[s * (HDD / 2) + p];
  float sn = fsin[s * (HDD / 2) + p];
  float x0 = (float)x[base], x1 = (float)x[base + 1];
  x[base]     = (__bf16)(x0 * c - x1 * sn);
  x[base + 1] = (__bf16)(x0 * sn + x1 * c);
}

// ---------------------------------------------------------------------------
// Flash attention (causal, GQA). grid = (B*H, S/128), block = 256 (8 waves).
// Each wave owns 16 query rows; all 8 waves share one 32-key K/V LDS tile.
// Scores computed TRANSPOSED (S^T = K * Q^T) so lane == query column ->
// softmax over keys is a pure per-lane reduction.
// O^T = V^T * P^T accumulated in 8 f32 C-fragments (d=128 -> 8 x 16 rows).
// ---------------------------------------------------------------------------
__global__ __launch_bounds__(256) void gqa_flash(
    const __bf16* __restrict__ qh, const __bf16* __restrict__ kh,
    const __bf16* __restrict__ vh, __bf16* __restrict__ outp)
{
  __shared__ unsigned short Kt[32 * HDD];  // [k][d] bf16 bits, 8 KB
  __shared__ unsigned short Vt[HDD * 32];  // [d][k] bf16 bits, 8 KB

  const int bh  = blockIdx.x;
  const int b   = bh / HH;
  const int h   = bh % HH;
  const int kvh = h / GROUPP;
  const int lane = threadIdx.x & 31;
  const int wave = threadIdx.x >> 5;
  const bool lo  = lane < 16;
  const int  lq  = lane & 15;
  const int qb_block = blockIdx.y * 128;
  const int q_abs = qb_block + wave * 16 + lq;   // this lane's query row

  // Q^T B-operand fragments for the 4 K-chunks of d (preloaded, reused all iters)
  const __bf16* qrow = qh + ((size_t)(b * SS + q_abs)) * DD + h * HDD;
  v16bf qf[4];
#pragma unroll
  for (int dk = 0; dk < 4; ++dk)
    qf[dk] = load_b_frag_bf(qrow + dk * 32 + (lo ? 0 : 16));

  v8f  o[8];
#pragma unroll
  for (int dc = 0; dc < 8; ++dc) o[dc] = (v8f){};
  float mrun = -1e30f, lsum = 0.f;

  const int nit = blockIdx.y * 4 + 4;            // (qb_block+128)/32 key tiles
  for (int kk = 0; kk < nit; ++kk) {
    const int k_base = kk * 32;

    // --- stage K (row-major) and V^T tiles: thread (kr=tid&31, dh=tid>>5)
    // copies a 16-wide d slice of key row kr.
    {
      const int kr = threadIdx.x & 31;
      const int dh = threadIdx.x >> 5;           // 0..7 -> d-chunk of 16
      const size_t rbase = ((size_t)(b * SS + k_base + kr)) * (KVHH * HDD)
                           + kvh * HDD + dh * 16;
      bf16x8 ku0 = *(const bf16x8*)(kh + rbase);
      bf16x8 ku1 = *(const bf16x8*)(kh + rbase + 8);
      *(bf16x8*)(&Kt[kr * HDD + dh * 16])     = ku0;
      *(bf16x8*)(&Kt[kr * HDD + dh * 16 + 8]) = ku1;
      bf16x8 vu0 = *(const bf16x8*)(vh + rbase);
      bf16x8 vu1 = *(const bf16x8*)(vh + rbase + 8);
#pragma unroll
      for (int i = 0; i < 8; ++i) {
        Vt[(dh * 16 + i) * 32 + kr]     = __builtin_bit_cast(unsigned short, vu0[i]);
        Vt[(dh * 16 + 8 + i) * 32 + kr] = __builtin_bit_cast(unsigned short, vu1[i]);
      }
    }
    __syncthreads();

    // --- S^T tiles: s0 = keys k_base..+15, s1 = keys +16..+31 (A = K rows, LDS)
    v8f s0 = {}; v8f s1 = {};
#pragma unroll
    for (int dk = 0; dk < 4; ++dk) {
      v16bf k0 = load_a_frag_lds(&Kt[lq * HDD + dk * 32], lo);
      v16bf k1 = load_a_frag_lds(&Kt[(16 + lq) * HDD + dk * 32], lo);
      s0 = WMMA_BF16(k0, qf[dk], s0);
      s1 = WMMA_BF16(k1, qf[dk], s1);
    }

    // --- per-lane causal mask + online softmax over the 16 key values
    float sv0[8], sv1[8], tmax = -1e30f;
#pragma unroll
    for (int j = 0; j < 8; ++j) {
      int kr0 = k_base + (lo ? j : j + 8);
      int kr1 = kr0 + 16;
      sv0[j] = (kr0 <= q_abs) ? s0[j] * SCALE_QK : -1e30f;
      sv1[j] = (kr1 <= q_abs) ? s1[j] * SCALE_QK : -1e30f;
      tmax = fmaxf(tmax, fmaxf(sv0[j], sv1[j]));
    }
    float mnew  = fmaxf(mrun, tmax);
    float alpha = __expf(mrun - mnew);
    float p0[8], p1[8], psum = 0.f;
#pragma unroll
    for (int j = 0; j < 8; ++j) {
      p0[j] = __expf(sv0[j] - mnew);
      p1[j] = __expf(sv1[j] - mnew);
      psum += p0[j] + p1[j];
    }
    lsum = lsum * alpha + psum;
    mrun = mnew;
#pragma unroll
    for (int dc = 0; dc < 8; ++dc)
#pragma unroll
      for (int j = 0; j < 8; ++j) o[dc][j] *= alpha;

    // --- P^T as B operand: lane half-swap via shfl_xor(16)
    v16bf pf;
#pragma unroll
    for (int j = 0; j < 8; ++j) {
      float x0 = __shfl_xor(p0[j], 16, 32);
      float x1 = __shfl_xor(p1[j], 16, 32);
      pf[j]     = (__bf16)(lo ? p0[j] : x1);
      pf[j + 8] = (__bf16)(lo ? x0 : p1[j]);
    }

    // --- O^T += V^T (A from LDS) * P^T
#pragma unroll
    for (int dc = 0; dc < 8; ++dc) {
      v16bf vf = load_a_frag_lds(&Vt[(dc * 16 + lq) * 32], lo);
      o[dc] = WMMA_BF16(vf, pf, o[dc]);
    }
    __syncthreads();
  }

  // --- normalize, convert to bf16, store packed 16-B per fragment
  const float rl = 1.0f / lsum;
  __bf16* orow = outp + ((size_t)(b * SS + q_abs)) * DD + h * HDD;
#pragma unroll
  for (int dc = 0; dc < 8; ++dc) {
    const int dbase = dc * 16 + (lo ? 0 : 8);
    bf16x8 w;
#pragma unroll
    for (int i = 0; i < 8; ++i) w[i] = (__bf16)(o[dc][i] * rl);
    *(bf16x8*)(orow + dbase) = w;
  }
}

// ---------------------------------------------------------------------------
extern "C" void kernel_launch(void* const* d_in, const int* in_sizes, int n_in,
                              void* d_out, int out_size, void* d_ws, size_t ws_size,
                              hipStream_t stream) {
  const float* q  = (const float*)d_in[0];
  const float* k  = (const float*)d_in[1];
  const float* v  = (const float*)d_in[2];
  const float* Wq = (const float*)d_in[3];
  const float* Wk = (const float*)d_in[4];
  const float* Wv = (const float*)d_in[5];
  const float* Wo = (const float*)d_in[6];
  const float* fc = (const float*)d_in[7];
  const float* fs = (const float*)d_in[8];
  float* out = (float*)d_out;

  char* ws = (char*)d_ws;
  // bf16 copies of inputs/weights (converted once, reused many times)
  __bf16* qbf  = (__bf16*)(ws);                       // 2048x4096 = 16 MB
  __bf16* kbf  = (__bf16*)(ws + (size_t)16777216);    // 16 MB
  __bf16* vbf  = (__bf16*)(ws + (size_t)33554432);    // 16 MB
  __bf16* Wqb  = (__bf16*)(ws + (size_t)50331648);    // 4096x4096 = 32 MB
  __bf16* Wkb  = (__bf16*)(ws + (size_t)83886080);    // 1024x4096 =  8 MB
  __bf16* Wvb  = (__bf16*)(ws + (size_t)92274688);    //  8 MB
  __bf16* Wob  = (__bf16*)(ws + (size_t)100663296);   // 32 MB
  // bf16 intermediates
  __bf16* qhb  = (__bf16*)(ws + (size_t)134217728);   // 2048x4096 = 16 MB
  __bf16* khb  = (__bf16*)(ws + (size_t)150994944);   // 2048x1024 =  4 MB
  __bf16* vhb  = (__bf16*)(ws + (size_t)155189248);   //  4 MB
  __bf16* attn = (__bf16*)(ws + (size_t)159383552);   // 16 MB

  const int M = BB * SS;                              // 2048
  const int NQ = DD, NKV = KVHH * HDD;                // 4096, 1024

  // one-time conversions
  auto cvt = [&](const float* s, __bf16* d, int n) {
    gqa_cvt_bf16<<<(unsigned)((n / 8 + 255) / 256), 256, 0, stream>>>(s, d, n);
  };
  cvt(q,  qbf, M * DD);
  cvt(k,  kbf, M * DD);
  cvt(v,  vbf, M * DD);
  cvt(Wq, Wqb, DD * DD);
  cvt(Wk, Wkb, NKV * DD);
  cvt(Wv, Wvb, NKV * DD);
  cvt(Wo, Wob, DD * DD);

  dim3 blk(256);
  gqa_gemm_wt<__bf16><<<dim3(NQ / 64,  M / 128), blk, 0, stream>>>(qbf, Wqb, qhb, M, NQ,  DD);
  gqa_gemm_wt<__bf16><<<dim3(NKV / 64, M / 128), blk, 0, stream>>>(kbf, Wkb, khb, M, NKV, DD);
  gqa_gemm_wt<__bf16><<<dim3(NKV / 64, M / 128), blk, 0, stream>>>(vbf, Wvb, vhb, M, NKV, DD);

  {
    size_t tq = (size_t)BB * SS * HH * (HDD / 2);
    gqa_rope<<<(unsigned)((tq + 255) / 256), 256, 0, stream>>>(qhb, fc, fs, HH);
    size_t tk = (size_t)BB * SS * KVHH * (HDD / 2);
    gqa_rope<<<(unsigned)((tk + 255) / 256), 256, 0, stream>>>(khb, fc, fs, KVHH);
  }

  gqa_flash<<<dim3(BB * HH, SS / 128), dim3(256), 0, stream>>>(qhb, khb, vhb, attn);

  gqa_gemm_wt<float><<<dim3(DD / 64, M / 128), blk, 0, stream>>>(attn, Wob, out, M, DD, DD);
}
